// LinkPredictionHead_62053687493367
// MI455X (gfx1250) — matrix operand order repaired and checked by reading, not compiled, not run on hardware
//
#include <hip/hip_runtime.h>
#include <hip/hip_bf16.h>
#include <math.h>

#define E_EDGES 500000
#define EMB_D   256
#define K_TOT   512        // 2*D
#define MB      128        // edges per block (8 M-tiles of 16)
#define KB_TOT  16         // 512 / 32

typedef __attribute__((ext_vector_type(16))) __bf16 bf16x16;
typedef __attribute__((ext_vector_type(8)))  float  f32x8;

union FragAB { bf16x16 v; uint4 q[2]; };

__device__ __forceinline__ unsigned int f2bf1(float f) {
    unsigned int u = __float_as_uint(f);
    return (u + (((u >> 16) & 1u) + 0x7fffu)) >> 16;   // round-to-nearest-even
}
__device__ __forceinline__ unsigned int packbf(float lo, float hi) {
    return f2bf1(lo) | (f2bf1(hi) << 16);
}

// ---------------------------------------------------------------------------
// Prep: repack W1 [512,256] f32 row-major -> bf16 B-fragment layout in d_ws.
// dword index i: j = i&7 (vgpr), lane = (i>>3)&31, kb = (i>>8)&15, nt = i>>12.
// Lane L of B-frag (nt,kb) holds n = nt*16 + L%16, K = kb*32 + 16*(L/16) + 2j
// packed (even K in [15:0], odd K in [31:16]).
// ---------------------------------------------------------------------------
__global__ void prep_w1_bf16(const float* __restrict__ W1,
                             unsigned int* __restrict__ bbuf) {
    int i    = blockIdx.x * blockDim.x + threadIdx.x;   // 0 .. 65535
    int j    = i & 7;
    int lane = (i >> 3) & 31;
    int kb   = (i >> 8) & 15;
    int nt   = i >> 12;
    int n    = nt * 16 + (lane & 15);
    int k    = kb * 32 + ((lane >> 4) << 4) + 2 * j;
    bbuf[i]  = packbf(W1[k * EMB_D + n], W1[(k + 1) * EMB_D + n]);
}

// ---------------------------------------------------------------------------
// Main: gather -> LDS bf16, 16-step K loop of v_wmma_f32_16x16x32_bf16,
// fused relu + dot(W2) reduction via ds_add_f32, sigmoid store.
// ---------------------------------------------------------------------------
__global__ __launch_bounds__(256, 1)
void edge_mlp_kernel(const float* __restrict__ emb,
                     const long long* __restrict__ eidx,
                     const float* __restrict__ b1,
                     const float* __restrict__ W2,
                     const float* __restrict__ b2,
                     const unsigned int* __restrict__ bbuf,
                     float* __restrict__ out) {
    extern __shared__ char smem[];                 // MB*512 bf16 (128KB) + MB floats
    float* accL = (float*)(smem + MB * K_TOT * 2);

    const int tid  = threadIdx.x;
    const int wave = tid >> 5;
    const int lane = tid & 31;
    const long long eBase = (long long)blockIdx.x * MB;

    if (tid < MB) accL[tid] = 0.0f;

    // ---- gather 128 edges x 512 f32 -> bf16 LDS (row-major, 1024 B/row) ----
    {
        int el    = tid >> 1;           // edge within block
        int which = tid & 1;            // 0 = src endpoint, 1 = dst endpoint
        long long e = eBase + el;
        if (e >= E_EDGES) e = 0;        // clamp; tail rows discarded at store
        long long node = eidx[which ? (E_EDGES + e) : e];
        const float4* src = (const float4*)(emb + node * EMB_D);     // 64 x b128
        uint4* dst = (uint4*)(smem + el * (K_TOT * 2) + which * EMB_D * 2);
#pragma unroll 4
        for (int f = 0; f < 32; ++f) {
            float4 a = src[2 * f];
            float4 b = src[2 * f + 1];
            uint4 p;
            p.x = packbf(a.x, a.y);
            p.y = packbf(a.z, a.w);
            p.z = packbf(b.x, b.y);
            p.w = packbf(b.z, b.w);
            dst[f] = p;
        }
    }
    __syncthreads();

    // ---- GEMM: 8 M-tiles x 2 N-tiles per wave, K = 512 in 16 steps ----
    f32x8 acc[8][2];
    {
        f32x8 z = {};
#pragma unroll
        for (int mt = 0; mt < 8; ++mt) { acc[mt][0] = z; acc[mt][1] = z; }
    }
    const int nt0 = wave * 2;                       // this wave's first N-tile

    // Per-M-tile LDS base pointers (constant across K loop) so the K loop
    // only uses small immediate ds offsets (kb*64, +16) -> no address VALU,
    // no WAR hazards on WMMA source regs.
    const char* aptr[8];
    {
        const int arow  = (lane & 15) * (K_TOT * 2);  // row-m byte offset in tile
        const int ahalf = (lane >> 4) * 32;           // K-half select (bytes)
#pragma unroll
        for (int mt = 0; mt < 8; ++mt)
            aptr[mt] = smem + mt * 16 * (K_TOT * 2) + arow + ahalf;
    }
    // Per-N-tile global base pointers; K loop uses kb*1024 immediate offsets.
    const uint4* bptr0 = (const uint4*)bbuf + ((((nt0    ) * 16) * 32 + lane) << 1);
    const uint4* bptr1 = (const uint4*)bbuf + ((((nt0 + 1) * 16) * 32 + lane) << 1);

#pragma unroll
    for (int kb = 0; kb < KB_TOT; ++kb) {
        // B fragments for this K step (2 x 32B per lane, from L2-resident W1)
        FragAB bf0, bf1;
        bf0.q[0] = bptr0[kb * 64];
        bf0.q[1] = bptr0[kb * 64 + 1];
        bf1.q[0] = bptr1[kb * 64];
        bf1.q[1] = bptr1[kb * 64 + 1];

        // All 8 A fragments into distinct registers -> one ds clause,
        // latency hidden across the 16 WMMAs below.
        FragAB a[8];
#pragma unroll
        for (int mt = 0; mt < 8; ++mt) {
            a[mt].q[0] = *(const uint4*)(aptr[mt] + kb * 64);
            a[mt].q[1] = *(const uint4*)(aptr[mt] + kb * 64 + 16);  // K + 8
        }
#pragma unroll
        for (int mt = 0; mt < 8; ++mt) {
            acc[mt][0] = __builtin_amdgcn_wmma_f32_16x16x32_bf16(
                false, a[mt].v, false, bf0.v, (short)0, acc[mt][0], false, false);
            acc[mt][1] = __builtin_amdgcn_wmma_f32_16x16x32_bf16(
                false, a[mt].v, false, bf1.v, (short)0, acc[mt][1], false, false);
        }
    }

    // ---- fused layer 2: relu(d + b1) * W2, reduce over n via LDS atomics ----
    const int nA = nt0 * 16 + (lane & 15);
    const int nB = (nt0 + 1) * 16 + (lane & 15);
    const float b1A = b1[nA], b1B = b1[nB];
    const float w2A = W2[nA], w2B = W2[nB];
    const int rofs = (lane >= 16) ? 8 : 0;
#pragma unroll
    for (int mt = 0; mt < 8; ++mt) {
#pragma unroll
        for (int r = 0; r < 8; ++r) {
            float h0 = acc[mt][0][r] + b1A; h0 = h0 > 0.0f ? h0 : 0.0f;
            float h1 = acc[mt][1][r] + b1B; h1 = h1 > 0.0f ? h1 : 0.0f;
            atomicAdd(&accL[mt * 16 + r + rofs], h0 * w2A + h1 * w2B);
        }
    }
    __syncthreads();

    if (tid < MB) {
        long long e = eBase + tid;
        if (e < E_EDGES) {
            float z = accL[tid] + b2[0];
            out[e] = 1.0f / (1.0f + expf(-z));
        }
    }
}

extern "C" void kernel_launch(void* const* d_in, const int* in_sizes, int n_in,
                              void* d_out, int out_size, void* d_ws, size_t ws_size,
                              hipStream_t stream) {
    const float*     emb  = (const float*)d_in[0];       // [100000, 256] f32
    const long long* eidx = (const long long*)d_in[1];   // [2, 500000] i64
    const float*     W1   = (const float*)d_in[2];       // [512, 256] f32
    const float*     b1   = (const float*)d_in[3];       // [256] f32
    const float*     W2   = (const float*)d_in[4];       // [256, 1] f32
    const float*     b2   = (const float*)d_in[5];       // [1] f32
    float*           out  = (float*)d_out;               // [500000] f32

    unsigned int* bbuf = (unsigned int*)d_ws;            // 256 KB bf16-packed W1

    // Repack W1 into per-lane B-fragment order (65536 dwords).
    prep_w1_bf16<<<256, 256, 0, stream>>>(W1, bbuf);

    const int nblocks = (E_EDGES + MB - 1) / MB;         // 3907
    const size_t shmem = (size_t)MB * K_TOT * 2 + MB * sizeof(float);
    edge_mlp_kernel<<<nblocks, 256, shmem, stream>>>(emb, eidx, b1, W2, b2, bbuf, out);
}
